// TransformerEncoderBlock_58841051955505
// MI455X (gfx1250) — matrix operand order repaired and checked by reading, not compiled, not run on hardware
//
#include <hip/hip_runtime.h>
#include <hip/hip_bf16.h>

// ---------------------------------------------------------------------------
// MI455X (gfx1250) transformer encoder block, bf16 WMMA + f32 accumulate.
// wave32, WMMA 16x16x32 bf16, flash attention, fused epilogues.
// ---------------------------------------------------------------------------

typedef __attribute__((ext_vector_type(16))) __bf16 bf16x16;
typedef __attribute__((ext_vector_type(8)))  __bf16 bf16x8;
typedef __attribute__((ext_vector_type(4)))  __bf16 bf16x4;
typedef __attribute__((ext_vector_type(8)))  float  f32x8;

#define EMBED   1024
#define HEADS   16
#define HDIM    64
#define HIDDEN  4096
#define SEQ     2048
#define ROWS    4096   // batch(2) * seq(2048)

__device__ __forceinline__ bf16x16 cat8(bf16x8 lo, bf16x8 hi) {
  return __builtin_shufflevector(lo, hi, 0,1,2,3,4,5,6,7,8,9,10,11,12,13,14,15);
}
__device__ __forceinline__ f32x8 wmma_bf16(bf16x16 a, bf16x16 b, f32x8 c) {
  // D = A(16x32 bf16) x B(32x16 bf16) + C(16x16 f32)
  return __builtin_amdgcn_wmma_f32_16x16x32_bf16(false, a, false, b,
                                                 (short)0, c, false, false);
}
__device__ __forceinline__ float gelu_f(float x) {
  // tanh-approx GELU (matches jax.nn.gelu default)
  float x3 = x * x * x;
  return 0.5f * x * (1.f + tanhf(0.7978845608028654f * (x + 0.044715f * x3)));
}

// ---------------------------------------------------------------------------
// f32 -> bf16 conversion (weights)
// ---------------------------------------------------------------------------
__global__ void k_cvt_bf16(const float* __restrict__ in,
                           __bf16* __restrict__ out, int n) {
  int i = (blockIdx.x * blockDim.x + threadIdx.x) * 4;
  if (i < n) {
    float4 v = *(const float4*)(in + i);
    bf16x4 o;
    o[0] = (__bf16)v.x; o[1] = (__bf16)v.y; o[2] = (__bf16)v.z; o[3] = (__bf16)v.w;
    *(bf16x4*)(out + i) = o;
  }
}

// ---------------------------------------------------------------------------
// LayerNorm over EMBED=1024, one block (256 thr) per row, bf16 output
// ---------------------------------------------------------------------------
__global__ __launch_bounds__(256) void k_layernorm_bf16(
    const float* __restrict__ x, const float* __restrict__ g,
    const float* __restrict__ b, __bf16* __restrict__ out) {
  __shared__ float rs[256], rq[256];
  const int tid = threadIdx.x;
  const int row = blockIdx.x;
  const float* xr = x + (size_t)row * EMBED;
  float4 v = *(const float4*)(xr + tid * 4);
  rs[tid] = v.x + v.y + v.z + v.w;
  rq[tid] = v.x*v.x + v.y*v.y + v.z*v.z + v.w*v.w;
  __syncthreads();
  for (int st = 128; st > 0; st >>= 1) {
    if (tid < st) { rs[tid] += rs[tid + st]; rq[tid] += rq[tid + st]; }
    __syncthreads();
  }
  float mu  = rs[0] * (1.f / EMBED);
  float var = rq[0] * (1.f / EMBED) - mu * mu;
  float inv = rsqrtf(var + 1e-5f);
  int c = tid * 4;
  bf16x4 o;
  o[0] = (__bf16)((v.x - mu) * inv * g[c + 0] + b[c + 0]);
  o[1] = (__bf16)((v.y - mu) * inv * g[c + 1] + b[c + 1]);
  o[2] = (__bf16)((v.z - mu) * inv * g[c + 2] + b[c + 2]);
  o[3] = (__bf16)((v.w - mu) * inv * g[c + 3] + b[c + 3]);
  *(bf16x4*)(out + (size_t)row * EMBED + c) = o;
}

// ---------------------------------------------------------------------------
// Tiled WMMA GEMM: C[M,N] = A[M,K](bf16) x B[K,N](bf16) + bias, epilogue:
//   EPI=0: out bf16            EPI=1: out f32 = residual + C
//   EPI=2: out bf16 = gelu(C)
// Block 256 thr = 8 waves, 128x128 output tile, K-tile 32.
// Wave (wm 0..3, wn 0..1) owns 32x64 = 2x4 frags of 16x16.
// ---------------------------------------------------------------------------
template <int EPI>
__global__ __launch_bounds__(256) void k_gemm_bf16(
    const __bf16* __restrict__ A, const __bf16* __restrict__ B,
    const float* __restrict__ bias, const float* __restrict__ residual,
    float* __restrict__ outF, __bf16* __restrict__ outB,
    int M, int N, int K) {
  __shared__ __bf16 sA[128 * 40];   // [m][k] row stride 40 (pad, 16B aligned)
  __shared__ __bf16 sBt[128 * 40];  // transposed: [n][k] stride 40

  const int tid  = threadIdx.x;
  const int lane = tid & 31;
  const int w    = tid >> 5;
  const int wm   = w >> 1, wn = w & 1;
  const int hl   = lane >> 4;       // half-wave (K/M split per ISA layout)
  const int l16  = lane & 15;
  const int rowBase = blockIdx.y * 128, colBase = blockIdx.x * 128;

  f32x8 acc[2][4];
#pragma unroll
  for (int tm = 0; tm < 2; tm++)
#pragma unroll
    for (int tn = 0; tn < 4; tn++)
#pragma unroll
      for (int e = 0; e < 8; e++) acc[tm][tn][e] = 0.f;

  const int rA = tid >> 1, kA = (tid & 1) * 16;       // A: 16 elems / thread
  const int kB = tid >> 3, nB = (tid & 7) * 16;       // B: 16 elems / thread
  const __bf16* aptr = A + (size_t)(rowBase + rA) * K + kA;
  const __bf16* bptr = B + (size_t)kB * N + colBase + nB;

  for (int k0 = 0; k0 < K; k0 += 32) {
    __syncthreads();
    // stage A tile (row-major)
    bf16x8 a0 = *(const bf16x8*)(aptr + k0);
    bf16x8 a1 = *(const bf16x8*)(aptr + k0 + 8);
    *(bf16x8*)&sA[rA * 40 + kA]     = a0;
    *(bf16x8*)&sA[rA * 40 + kA + 8] = a1;
    // stage B tile transposed -> per-lane-contiguous B fragments
    const __bf16* bp = bptr + (size_t)k0 * N;
    bf16x8 b0 = *(const bf16x8*)(bp);
    bf16x8 b1 = *(const bf16x8*)(bp + 8);
#pragma unroll
    for (int e = 0; e < 8; e++) sBt[(nB + e) * 40 + kB]     = b0[e];
#pragma unroll
    for (int e = 0; e < 8; e++) sBt[(nB + 8 + e) * 40 + kB] = b1[e];
    // prefetch next K-tile (global_prefetch_b8, speculative ok on last iter)
    __builtin_prefetch(aptr + k0 + 32, 0, 0);
    __builtin_prefetch(bp + (size_t)32 * N, 0, 0);
    __syncthreads();

    bf16x16 bfr[4];
#pragma unroll
    for (int tn = 0; tn < 4; tn++) {
      int n = wn * 64 + tn * 16 + l16;            // B frag: col = lane%16
      bf16x8 lo = *(const bf16x8*)&sBt[n * 40 + hl * 8];       // k = hl*8+j
      bf16x8 hi = *(const bf16x8*)&sBt[n * 40 + 16 + hl * 8];
      bfr[tn] = cat8(lo, hi);
    }
#pragma unroll
    for (int tm = 0; tm < 2; tm++) {
      int m = wm * 32 + tm * 16 + l16;            // A frag: row = lane%16
      bf16x8 lo = *(const bf16x8*)&sA[m * 40 + hl * 8];        // k = hl*8+j
      bf16x8 hi = *(const bf16x8*)&sA[m * 40 + 16 + hl * 8];   // k = 16+hl*8+j
      bf16x16 af = cat8(lo, hi);
#pragma unroll
      for (int tn = 0; tn < 4; tn++)
        acc[tm][tn] = wmma_bf16(af, bfr[tn], acc[tm][tn]);
    }
  }

  // epilogue: C layout row = j + 8*hl, col = lane%16
#pragma unroll
  for (int tm = 0; tm < 2; tm++) {
#pragma unroll
    for (int tn = 0; tn < 4; tn++) {
      int col = colBase + wn * 64 + tn * 16 + l16;
      float bv = bias[col];
#pragma unroll
      for (int j = 0; j < 8; j++) {
        int row = rowBase + wm * 32 + tm * 16 + hl * 8 + j;
        size_t idx = (size_t)row * N + col;
        float v = acc[tm][tn][j] + bv;
        if (EPI == 0)      outB[idx] = (__bf16)v;
        else if (EPI == 1) outF[idx] = residual[idx] + v;
        else               outB[idx] = (__bf16)gelu_f(v);
      }
    }
  }
}

// ---------------------------------------------------------------------------
// Flash attention. qkv layout: [4096 rows][3072] = [.., 3, H, D] bf16.
// Grid: (qtile 0..15, batch*head 0..31). Block 256 thr = 8 waves.
// Wave w owns 16 query rows; online softmax; O accum in f32 frags.
// ---------------------------------------------------------------------------
__global__ __launch_bounds__(256) void k_attention(
    const __bf16* __restrict__ qkv, __bf16* __restrict__ out) {
  __shared__ __bf16 sK [128 * 64];       // [key][d] row-major
  __shared__ __bf16 sVt[64 * 128];       // [d][key] transposed
  __shared__ __bf16 sP [8 * 16 * 128];   // per-wave P strip [16 q][128 key]

  const int tid  = threadIdx.x;
  const int lane = tid & 31;
  const int w    = tid >> 5;
  const int hl   = lane >> 4, l16 = lane & 15;
  const int bh = blockIdx.y, bb = bh >> 4, h = bh & 15;
  const int q0 = blockIdx.x * 128;
  const size_t rowoff = (size_t)bb * SEQ;

  // Q fragments held in registers across whole KV loop (row = lane%16)
  const int qrow = q0 + w * 16 + l16;
  const __bf16* qptr = qkv + (rowoff + qrow) * 3072 + h * 64;  // s=0 (Q)
  bf16x16 qf[2];
#pragma unroll
  for (int ks = 0; ks < 2; ks++) {
    bf16x8 lo = *(const bf16x8*)(qptr + ks * 32 + hl * 8);
    bf16x8 hi = *(const bf16x8*)(qptr + ks * 32 + 16 + hl * 8);
    qf[ks] = cat8(lo, hi);
  }

  f32x8 o[4];
  float mrow[8], lrow[8];
#pragma unroll
  for (int f = 0; f < 4; f++)
#pragma unroll
    for (int e = 0; e < 8; e++) o[f][e] = 0.f;
#pragma unroll
  for (int j = 0; j < 8; j++) { mrow[j] = -3.0e38f; lrow[j] = 0.f; }

  __bf16* pw = sP + w * (16 * 128);

  for (int kv0 = 0; kv0 < SEQ; kv0 += 128) {
    __syncthreads();
    {  // cooperative stage of K (row-major) and V (transposed)
      int key = tid >> 1, d0 = (tid & 1) * 32;
      const __bf16* kp = qkv + (rowoff + kv0 + key) * 3072 + 1024 + h * 64 + d0;
      const __bf16* vp = qkv + (rowoff + kv0 + key) * 3072 + 2048 + h * 64 + d0;
#pragma unroll
      for (int i = 0; i < 4; i++) {
        bf16x8 kv_ = *(const bf16x8*)(kp + i * 8);
        *(bf16x8*)&sK[key * 64 + d0 + i * 8] = kv_;
        bf16x8 vv = *(const bf16x8*)(vp + i * 8);
#pragma unroll
        for (int e = 0; e < 8; e++) sVt[(d0 + i * 8 + e) * 128 + key] = vv[e];
      }
    }
    __syncthreads();

    // S = Q x K^T  (16 x 128 per wave, 8 frags, contraction over d=64)
    f32x8 s[8];
#pragma unroll
    for (int t = 0; t < 8; t++)
#pragma unroll
      for (int e = 0; e < 8; e++) s[t][e] = 0.f;
#pragma unroll
    for (int ks = 0; ks < 2; ks++) {
#pragma unroll
      for (int t = 0; t < 8; t++) {
        int keyl = t * 16 + l16;  // B frag col = key
        bf16x8 lo = *(const bf16x8*)&sK[keyl * 64 + ks * 32 + hl * 8];
        bf16x8 hi = *(const bf16x8*)&sK[keyl * 64 + ks * 32 + 16 + hl * 8];
        s[t] = wmma_bf16(qf[ks], cat8(lo, hi), s[t]);
      }
    }

    // online softmax (scale 1/sqrt(64)=0.125); reductions over 16-lane half
#pragma unroll
    for (int t = 0; t < 8; t++)
#pragma unroll
      for (int j = 0; j < 8; j++) s[t][j] *= 0.125f;
#pragma unroll
    for (int j = 0; j < 8; j++) {
      float v = -3.0e38f;
#pragma unroll
      for (int t = 0; t < 8; t++) v = fmaxf(v, s[t][j]);
#pragma unroll
      for (int off = 1; off < 16; off <<= 1)
        v = fmaxf(v, __shfl_xor(v, off, 32));
      float mn = fmaxf(mrow[j], v);
      float alpha = __expf(mrow[j] - mn);
      mrow[j] = mn;
      float rsum = 0.f;
#pragma unroll
      for (int t = 0; t < 8; t++) {
        float p = __expf(s[t][j] - mn);
        s[t][j] = p;
        rsum += p;
      }
#pragma unroll
      for (int off = 1; off < 16; off <<= 1)
        rsum += __shfl_xor(rsum, off, 32);
      lrow[j] = lrow[j] * alpha + rsum;
#pragma unroll
      for (int f = 0; f < 4; f++) o[f][j] *= alpha;
    }

    // re-lay P: C-layout regs -> per-wave LDS strip -> A-layout frags
#pragma unroll
    for (int t = 0; t < 8; t++)
#pragma unroll
      for (int j = 0; j < 8; j++)
        pw[(hl * 8 + j) * 128 + t * 16 + l16] = (__bf16)s[t][j];

    // O += P x V  (contraction over 128 keys, 4 k-steps of 32)
#pragma unroll
    for (int kk = 0; kk < 4; kk++) {
      bf16x8 lo = *(const bf16x8*)&pw[l16 * 128 + kk * 32 + hl * 8];
      bf16x8 hi = *(const bf16x8*)&pw[l16 * 128 + kk * 32 + 16 + hl * 8];
      bf16x16 pa = cat8(lo, hi);
#pragma unroll
      for (int f = 0; f < 4; f++) {
        int d = f * 16 + l16;  // B frag col = d
        bf16x8 blo = *(const bf16x8*)&sVt[d * 128 + kk * 32 + hl * 8];
        bf16x8 bhi = *(const bf16x8*)&sVt[d * 128 + kk * 32 + 16 + hl * 8];
        o[f] = wmma_bf16(pa, cat8(blo, bhi), o[f]);
      }
    }
  }

  // normalize and store: out[row][h*64 + d], bf16
#pragma unroll
  for (int j = 0; j < 8; j++) {
    float inv = 1.f / lrow[j];
    int row = q0 + w * 16 + hl * 8 + j;
    size_t base = (rowoff + row) * EMBED + h * 64;
#pragma unroll
    for (int f = 0; f < 4; f++)
      out[base + f * 16 + l16] = (__bf16)(o[f][j] * inv);
  }
}

// ---------------------------------------------------------------------------
// Launch: cvt weights -> LN1 -> QKV -> attn -> proj(+x) -> LN2 -> FC1(gelu)
//         -> FC2(+x1) -> d_out.  Workspace ~126 MB.
// ---------------------------------------------------------------------------
extern "C" void kernel_launch(void* const* d_in, const int* in_sizes, int n_in,
                              void* d_out, int out_size, void* d_ws,
                              size_t ws_size, hipStream_t stream) {
  const float* x      = (const float*)d_in[0];
  const float* qkv_w  = (const float*)d_in[1];
  const float* qkv_b  = (const float*)d_in[2];
  const float* proj_w = (const float*)d_in[3];
  const float* proj_b = (const float*)d_in[4];
  const float* fc1_w  = (const float*)d_in[5];
  const float* fc1_b  = (const float*)d_in[6];
  const float* fc2_w  = (const float*)d_in[7];
  const float* fc2_b  = (const float*)d_in[8];
  const float* ln1_g  = (const float*)d_in[9];
  const float* ln1_b  = (const float*)d_in[10];
  const float* ln2_g  = (const float*)d_in[11];
  const float* ln2_b  = (const float*)d_in[12];
  float* outp = (float*)d_out;

  char* ws = (char*)d_ws;
  size_t off = 0;
  auto alloc = [&](size_t bytes) -> char* {
    char* p = ws + off;
    off += (bytes + 255) & ~(size_t)255;
    return p;
  };
  __bf16* wQKV  = (__bf16*)alloc((size_t)EMBED * 3072 * 2);
  __bf16* wPROJ = (__bf16*)alloc((size_t)EMBED * EMBED * 2);
  __bf16* wFC1  = (__bf16*)alloc((size_t)EMBED * HIDDEN * 2);
  __bf16* wFC2  = (__bf16*)alloc((size_t)HIDDEN * EMBED * 2);
  __bf16* ln1x  = (__bf16*)alloc((size_t)ROWS * EMBED * 2);
  __bf16* qkvb  = (__bf16*)alloc((size_t)ROWS * 3072 * 2);
  __bf16* attnb = (__bf16*)alloc((size_t)ROWS * EMBED * 2);
  float*  x1    = (float*) alloc((size_t)ROWS * EMBED * 4);
  __bf16* ln2x  = (__bf16*)alloc((size_t)ROWS * EMBED * 2);
  __bf16* hb    = (__bf16*)alloc((size_t)ROWS * HIDDEN * 2);

  auto cvt = [&](const float* src, __bf16* dst, int n) {
    k_cvt_bf16<<<dim3((n / 4 + 255) / 256), 256, 0, stream>>>(src, dst, n);
  };
  cvt(qkv_w,  wQKV,  EMBED * 3072);
  cvt(proj_w, wPROJ, EMBED * EMBED);
  cvt(fc1_w,  wFC1,  EMBED * HIDDEN);
  cvt(fc2_w,  wFC2,  HIDDEN * EMBED);

  // LN1
  k_layernorm_bf16<<<ROWS, 256, 0, stream>>>(x, ln1_g, ln1_b, ln1x);
  // QKV: [4096,3072] bf16
  k_gemm_bf16<0><<<dim3(3072 / 128, ROWS / 128), 256, 0, stream>>>(
      ln1x, wQKV, qkv_b, nullptr, nullptr, qkvb, ROWS, 3072, EMBED);
  // attention
  k_attention<<<dim3(SEQ / 128, 2 * HEADS), 256, 0, stream>>>(qkvb, attnb);
  // proj + residual(x) -> x1 (f32)
  k_gemm_bf16<1><<<dim3(EMBED / 128, ROWS / 128), 256, 0, stream>>>(
      attnb, wPROJ, proj_b, x, x1, nullptr, ROWS, EMBED, EMBED);
  // LN2
  k_layernorm_bf16<<<ROWS, 256, 0, stream>>>(x1, ln2_g, ln2_b, ln2x);
  // FC1 + GELU -> h (bf16)
  k_gemm_bf16<2><<<dim3(HIDDEN / 128, ROWS / 128), 256, 0, stream>>>(
      ln2x, wFC1, fc1_b, nullptr, nullptr, hb, ROWS, HIDDEN, EMBED);
  // FC2 + residual(x1) -> d_out (f32)
  k_gemm_bf16<1><<<dim3(EMBED / 128, ROWS / 128), 256, 0, stream>>>(
      hb, wFC2, fc2_b, x1, outp, nullptr, ROWS, EMBED, HIDDEN);
}
